// BasicGRUClassifier_88064009437922
// MI455X (gfx1250) — matrix operand order repaired and checked by reading, not compiled, not run on hardware
//
#include <hip/hip_runtime.h>
#include <hip/hip_bf16.h>

typedef __attribute__((ext_vector_type(8)))  float  v8f;
typedef __attribute__((ext_vector_type(8)))  __bf16 v8bf;
typedef __attribute__((ext_vector_type(16))) __bf16 v16bf;

#define HIDN   128
#define IN_CH  271
#define SEQ    281
#define NCLS   1854
#define BATCH  256

#define XPAD   288          // 271 padded up to multiple of 32
#define NCH_X  9            // 288/32
#define NCH_H  4            // 128/32
#define NCH0   13           // layer0 K-chunks (288 + 128)/32
#define NCH1   8            // layer1 K-chunks 256/32
#define NT_FC  116          // ceil(1854/16)
#define NCH_FC 4            // 128/32

// workspace offsets (in bf16 elements)
#define SZ_L0   (NCH0 * 8 * 512)      // 53248 per layer0 gate
#define SZ_L1   (NCH1 * 8 * 512)      // 32768 per layer1 gate
#define SZ_FC   (NT_FC * NCH_FC * 512)
#define OFF_WR0 0
#define OFF_WU0 (OFF_WR0 + SZ_L0)
#define OFF_WO0 (OFF_WU0 + SZ_L0)
#define OFF_WR1 (OFF_WO0 + SZ_L0)
#define OFF_WU1 (OFF_WR1 + SZ_L1)
#define OFF_WO1 (OFF_WU1 + SZ_L1)
#define OFF_WFC (OFF_WO1 + SZ_L1)
#define OFF_H1  (OFF_WFC + SZ_FC)

__device__ __forceinline__ __bf16 f2bf(float f) {
    unsigned u = __builtin_bit_cast(unsigned, f);
    unsigned r = (u + 0x7FFFu + ((u >> 16) & 1u)) >> 16;
    unsigned short s = (unsigned short)r;
    return __builtin_bit_cast(__bf16, s);
}

// Fast activations: v_exp_f32 + v_rcp_f32 (TRANS ops, co-execute with VALU)
// instead of the IEEE divide expansion / libm tanh.
__device__ __forceinline__ float sigmoidf(float x) {
    return __builtin_amdgcn_rcpf(1.0f + __expf(-x));
}
__device__ __forceinline__ float tanh_fast(float x) {
    return 1.0f - 2.0f * __builtin_amdgcn_rcpf(__expf(2.0f * x) + 1.0f);
}

// A-fragment from a row base already offset to the chunk: two contiguous 16B
// runs per the ISA 16-bit A layout (K = e + 8*half ; K = e + 8 + 8*half).
__device__ __forceinline__ v16bf lda(const __bf16* chunkbase, int hlf) {
    const v8bf lo = *(const v8bf*)(chunkbase + 8 * hlf);
    const v8bf hi = *(const v8bf*)(chunkbase + 16 + 8 * hlf);
    return __builtin_shufflevector(lo, hi, 0,1,2,3,4,5,6,7,8,9,10,11,12,13,14,15);
}

// B-fragment from pre-packed weights: 32 contiguous bytes per lane.
__device__ __forceinline__ v16bf ldb(const __bf16* tilebase, int kc, int lane) {
    return *(const v16bf*)(tilebase + ((size_t)kc * 32 + lane) * 16);
}

__device__ __forceinline__ v8f wmma_bf16(v16bf a, v16bf b, v8f c) {
    return __builtin_amdgcn_wmma_f32_16x16x32_bf16(false, a, false, b, (short)0, c,
                                                   false, false);
}

// ---------------------------------------------------------------------------
// Pack fp32 [K,N] weights into bf16 B-fragment order:
// dst[((nt*nch + kc)*32 + lane)*16 + e], K_in_chunk = (e&7) + 8*(lane>>4) + 16*(e>>3).
// l0map: layer-0 K remap (x rows 0..270, zero pad 271..287, h rows at 288..415)
// ---------------------------------------------------------------------------
__global__ void pack_weights(const float* __restrict__ W, __bf16* __restrict__ dst,
                             int ntiles, int nch, int Kreal, int Nreal, int Nstride,
                             int l0map) {
    int id = blockIdx.x * blockDim.x + threadIdx.x;
    int total = ntiles * nch * 512;
    if (id >= total) return;
    int e    = id & 15;
    int lane = (id >> 4) & 31;
    int kc   = (id >> 9) % nch;
    int nt   = id / (nch << 9);
    int kk   = kc * 32 + (e & 7) + 8 * (lane >> 4) + 16 * (e >> 3);
    int col  = nt * 16 + (lane & 15);
    int sk;
    if (l0map) sk = (kk < IN_CH) ? kk : ((kk < XPAD) ? -1 : kk - (XPAD - IN_CH));
    else       sk = (kk < Kreal) ? kk : -1;
    float v = 0.0f;
    if (sk >= 0 && col < Nreal) v = W[(size_t)sk * Nstride + col];
    dst[id] = f2bf(v);
}

// ---------------------------------------------------------------------------
// GRU: 16 blocks x 256 threads (8 waves). Block = 16 batch rows; wave = one
// 16-column tile of HID. Full T loop in-kernel, state in LDS.
// B fragments stream from L2 each step inside #pragma unroll 1 loops with a
// one-chunk-ahead register double buffer (prevents LICM hoist + scratch spill).
// ---------------------------------------------------------------------------
__global__ __launch_bounds__(256)
void gru_kernel(const float* __restrict__ X,
                const float* __restrict__ br0, const float* __restrict__ bu0,
                const float* __restrict__ bo0,
                const float* __restrict__ br1, const float* __restrict__ bu1,
                const float* __restrict__ bo1,
                const __bf16* __restrict__ wpack, __bf16* __restrict__ h1out) {
    __shared__ __bf16 xblk[16][XPAD];
    __shared__ __bf16 h0b[16][HIDN];
    __shared__ __bf16 rh0b[16][HIDN];
    __shared__ __bf16 h1b[16][HIDN];
    __shared__ __bf16 rh1b[16][HIDN];
    __shared__ float  h0f[16][HIDN];
    __shared__ float  h1f[16][HIDN];

    const int tid  = threadIdx.x;
    const int wave = tid >> 5;
    const int lane = tid & 31;
    const int hlf  = lane >> 4;
    const int lid  = lane & 15;
    const int b0   = blockIdx.x * 16;
    const int ncol = wave * 16 + lid;

    // x-staging decomposition: 16 threads per batch row, 18 columns each
    const int xm = tid >> 4;          // batch row within tile
    const int xc = tid & 15;          // starting column

    // per-wave packed-weight tile bases (this wave's 16 output columns)
    const __bf16* pr0 = wpack + OFF_WR0 + (size_t)wave * NCH0 * 512;
    const __bf16* pu0 = wpack + OFF_WU0 + (size_t)wave * NCH0 * 512;
    const __bf16* po0 = wpack + OFF_WO0 + (size_t)wave * NCH0 * 512;
    const __bf16* pr1 = wpack + OFF_WR1 + (size_t)wave * NCH1 * 512;
    const __bf16* pu1 = wpack + OFF_WU1 + (size_t)wave * NCH1 * 512;
    const __bf16* po1 = wpack + OFF_WO1 + (size_t)wave * NCH1 * 512;

    const float bias_r0 = br0[ncol], bias_u0 = bu0[ncol], bias_o0 = bo0[ncol];
    const float bias_r1 = br1[ncol], bias_u1 = bu1[ncol], bias_o1 = bo1[ncol];

    for (int i = tid; i < 16 * HIDN; i += 256) {
        int m = i >> 7, c = i & 127;
        h0f[m][c] = 0.0f;  h1f[m][c] = 0.0f;
        h0b[m][c] = f2bf(0.0f);  h1b[m][c] = f2bf(0.0f);
    }
    __syncthreads();

    for (int t = 0; t < SEQ; ++t) {
        // ---- stage x_t tile (strided gather; X is fully L2-resident) ----
        {
            const float* xrow = &X[(size_t)(b0 + xm) * IN_CH * SEQ + t];
            #pragma unroll
            for (int j = 0; j < XPAD / 16; ++j) {
                int c = xc + j * 16;
                float v = 0.0f;
                if (c < IN_CH) {
                    const float* p = xrow + (size_t)c * SEQ;
                    v = *p;
                    if (t + 16 < SEQ) __builtin_prefetch(p + 16, 0, 1);
                }
                xblk[xm][c] = f2bf(v);
            }
        }
        __syncthreads();

        // ---- layer 0: reset & update gates (K = [x(288) | h0(128)]) ----
        v8f ar, au;
        #pragma unroll
        for (int i = 0; i < 8; ++i) { ar[i] = bias_r0; au[i] = bias_u0; }
        {
            v16bf cbr = ldb(pr0, 0, lane);
            v16bf cbu = ldb(pu0, 0, lane);
            #pragma unroll 1
            for (int kc = 0; kc < NCH0; ++kc) {
                const __bf16* abase = (kc < NCH_X)
                    ? &xblk[lid][kc * 32] : &h0b[lid][(kc - NCH_X) * 32];
                int nk = (kc + 1 < NCH0) ? (kc + 1) : kc;
                v16bf nbr = ldb(pr0, nk, lane);          // prefetch next chunk
                v16bf nbu = ldb(pu0, nk, lane);
                v16bf a = lda(abase, hlf);
                ar = wmma_bf16(a, cbr, ar);
                au = wmma_bf16(a, cbu, au);
                cbr = nbr;  cbu = nbu;
            }
        }
        float uv0[8];
        #pragma unroll
        for (int r = 0; r < 8; ++r) {
            int row = r + 8 * hlf;
            float rv = sigmoidf(ar[r]);
            uv0[r]   = sigmoidf(au[r]);
            rh0b[row][ncol] = f2bf(rv * h0f[row][ncol]);
        }
        __syncthreads();

        // ---- layer 0: candidate (K = [x | r*h0]) + state update ----
        v8f ao;
        #pragma unroll
        for (int i = 0; i < 8; ++i) ao[i] = bias_o0;
        {
            v16bf cb = ldb(po0, 0, lane);
            #pragma unroll 1
            for (int kc = 0; kc < NCH0; ++kc) {
                const __bf16* abase = (kc < NCH_X)
                    ? &xblk[lid][kc * 32] : &rh0b[lid][(kc - NCH_X) * 32];
                int nk = (kc + 1 < NCH0) ? (kc + 1) : kc;
                v16bf nb = ldb(po0, nk, lane);
                ao = wmma_bf16(lda(abase, hlf), cb, ao);
                cb = nb;
            }
        }
        #pragma unroll
        for (int r = 0; r < 8; ++r) {
            int row = r + 8 * hlf;
            float o  = tanh_fast(ao[r]);
            float hn = (1.0f - uv0[r]) * h0f[row][ncol] + uv0[r] * o;
            h0f[row][ncol] = hn;
            h0b[row][ncol] = f2bf(hn);
        }
        __syncthreads();

        // ---- layer 1: reset & update gates (K = [h0_new(128) | h1(128)]) ----
        v8f ar1, au1;
        #pragma unroll
        for (int i = 0; i < 8; ++i) { ar1[i] = bias_r1; au1[i] = bias_u1; }
        {
            v16bf cbr = ldb(pr1, 0, lane);
            v16bf cbu = ldb(pu1, 0, lane);
            #pragma unroll 1
            for (int kc = 0; kc < NCH1; ++kc) {
                const __bf16* abase = (kc < NCH_H)
                    ? &h0b[lid][kc * 32] : &h1b[lid][(kc - NCH_H) * 32];
                int nk = (kc + 1 < NCH1) ? (kc + 1) : kc;
                v16bf nbr = ldb(pr1, nk, lane);
                v16bf nbu = ldb(pu1, nk, lane);
                v16bf a = lda(abase, hlf);
                ar1 = wmma_bf16(a, cbr, ar1);
                au1 = wmma_bf16(a, cbu, au1);
                cbr = nbr;  cbu = nbu;
            }
        }
        float uv1[8];
        #pragma unroll
        for (int r = 0; r < 8; ++r) {
            int row = r + 8 * hlf;
            float rv = sigmoidf(ar1[r]);
            uv1[r]   = sigmoidf(au1[r]);
            rh1b[row][ncol] = f2bf(rv * h1f[row][ncol]);
        }
        __syncthreads();

        // ---- layer 1: candidate (K = [h0_new | r1*h1]) + state update ----
        v8f ao1;
        #pragma unroll
        for (int i = 0; i < 8; ++i) ao1[i] = bias_o1;
        {
            v16bf cb = ldb(po1, 0, lane);
            #pragma unroll 1
            for (int kc = 0; kc < NCH1; ++kc) {
                const __bf16* abase = (kc < NCH_H)
                    ? &h0b[lid][kc * 32] : &rh1b[lid][(kc - NCH_H) * 32];
                int nk = (kc + 1 < NCH1) ? (kc + 1) : kc;
                v16bf nb = ldb(po1, nk, lane);
                ao1 = wmma_bf16(lda(abase, hlf), cb, ao1);
                cb = nb;
            }
        }
        #pragma unroll
        for (int r = 0; r < 8; ++r) {
            int row = r + 8 * hlf;
            float o  = tanh_fast(ao1[r]);
            float hn = (1.0f - uv1[r]) * h1f[row][ncol] + uv1[r] * o;
            h1f[row][ncol] = hn;
            h1b[row][ncol] = f2bf(hn);
        }
        __syncthreads();
    }

    // final h1 -> workspace (bf16 row-major [256][128])
    for (int i = tid; i < 16 * HIDN; i += 256) {
        int m = i >> 7, c = i & 127;
        h1out[(size_t)(b0 + m) * HIDN + c] = f2bf(h1f[m][c]);
    }
}

// ---------------------------------------------------------------------------
// Classifier: out[256,1854] = h1 @ Wfc + bfc.  One wave per 16x16 tile.
// ---------------------------------------------------------------------------
__global__ __launch_bounds__(32)
void fc_kernel(const __bf16* __restrict__ wfc_pack, const __bf16* __restrict__ h1,
               const float* __restrict__ bfc, float* __restrict__ out) {
    const int lane = threadIdx.x;
    const int hlf  = lane >> 4;
    const int lid  = lane & 15;
    const int mt   = blockIdx.x;      // 0..15
    const int nt   = blockIdx.y;      // 0..115
    const int col  = nt * 16 + lid;

    float bias = (col < NCLS) ? bfc[col] : 0.0f;
    v8f acc;
    #pragma unroll
    for (int i = 0; i < 8; ++i) acc[i] = bias;

    const __bf16* bp   = wfc_pack + (size_t)nt * NCH_FC * 512;
    const __bf16* arow = h1 + (size_t)(mt * 16 + lid) * HIDN;
    #pragma unroll
    for (int kc = 0; kc < NCH_FC; ++kc)
        acc = wmma_bf16(lda(arow + kc * 32, hlf), ldb(bp, kc, lane), acc);

    #pragma unroll
    for (int r = 0; r < 8; ++r) {
        int row = mt * 16 + r + 8 * hlf;
        if (col < NCLS) out[(size_t)row * NCLS + col] = acc[r];
    }
}

extern "C" void kernel_launch(void* const* d_in, const int* in_sizes, int n_in,
                              void* d_out, int out_size, void* d_ws, size_t ws_size,
                              hipStream_t stream) {
    const float* X   = (const float*)d_in[0];
    const float* Wr0 = (const float*)d_in[1];  const float* br0 = (const float*)d_in[2];
    const float* Wu0 = (const float*)d_in[3];  const float* bu0 = (const float*)d_in[4];
    const float* Wo0 = (const float*)d_in[5];  const float* bo0 = (const float*)d_in[6];
    const float* Wr1 = (const float*)d_in[7];  const float* br1 = (const float*)d_in[8];
    const float* Wu1 = (const float*)d_in[9];  const float* bu1 = (const float*)d_in[10];
    const float* Wo1 = (const float*)d_in[11]; const float* bo1 = (const float*)d_in[12];
    const float* Wfc = (const float*)d_in[13]; const float* bfc = (const float*)d_in[14];
    __bf16* ws  = (__bf16*)d_ws;
    float*  out = (float*)d_out;

    auto pack = [&](const float* W, size_t off, int ntiles, int nch, int Kreal,
                    int Nreal, int Nstride, int l0) {
        int total = ntiles * nch * 512;
        pack_weights<<<(total + 255) / 256, 256, 0, stream>>>(
            W, ws + off, ntiles, nch, Kreal, Nreal, Nstride, l0);
    };
    pack(Wr0, OFF_WR0, 8, NCH0, 0, 128, 128, 1);
    pack(Wu0, OFF_WU0, 8, NCH0, 0, 128, 128, 1);
    pack(Wo0, OFF_WO0, 8, NCH0, 0, 128, 128, 1);
    pack(Wr1, OFF_WR1, 8, NCH1, 256, 128, 128, 0);
    pack(Wu1, OFF_WU1, 8, NCH1, 256, 128, 128, 0);
    pack(Wo1, OFF_WO1, 8, NCH1, 256, 128, 128, 0);
    pack(Wfc, OFF_WFC, NT_FC, NCH_FC, 128, NCLS, NCLS, 0);

    gru_kernel<<<16, 256, 0, stream>>>(X, br0, bu0, bo0, br1, bu1, bo1,
                                       ws, ws + OFF_H1);

    dim3 g(16, NT_FC);
    fc_kernel<<<g, 32, 0, stream>>>(ws + OFF_WFC, ws + OFF_H1, bfc, out);
}
